// HeteroGNN_13657996002018
// MI455X (gfx1250) — compile-verified, matrix operand order checked by reading, not run on hardware
//
#include <hip/hip_runtime.h>

#define NP 4096
#define NF 16384
#define EPP 131072
#define EFF 262144
#define EFP 524288
#define ENC_NEG_INF 0x007FFFFFu  // encf(-inf)

typedef __attribute__((ext_vector_type(16))) _Float16 v16h;
typedef __attribute__((ext_vector_type(8)))  _Float16 v8h;
typedef __attribute__((ext_vector_type(8)))  float    v8f;

// ---------------- device helpers ----------------
__device__ __forceinline__ unsigned encf(float f) {
  unsigned u = __float_as_uint(f);
  return (u & 0x80000000u) ? ~u : (u | 0x80000000u);
}
__device__ __forceinline__ float decf(unsigned u) {
  return __uint_as_float((u & 0x80000000u) ? (u & 0x7FFFFFFFu) : ~u);
}

// ---------------- utility kernels ----------------
__global__ void fill_u32_k(unsigned* __restrict__ p, unsigned v, long n) {
  long i = (long)blockIdx.x * blockDim.x + threadIdx.x;
  if (i < n) p[i] = v;
}

__global__ void f32_to_f16_k(const float* __restrict__ s, _Float16* __restrict__ d, long n) {
  long i = (long)blockIdx.x * blockDim.x + threadIdx.x;
  if (i < n) d[i] = (_Float16)s[i];
}

// sum-reduce edge_attr columns (caller scales by 1/E later)
__global__ void mean_accum_k(const float* __restrict__ ea, float* __restrict__ out, int E, int edim) {
  float a0 = 0.f, a1 = 0.f;
  for (int e = blockIdx.x * blockDim.x + threadIdx.x; e < E; e += gridDim.x * blockDim.x) {
    a0 += ea[(size_t)e * edim];
    if (edim > 1) a1 += ea[(size_t)e * edim + 1];
  }
  for (int off = 16; off > 0; off >>= 1) {
    a0 += __shfl_xor(a0, off, 32);
    a1 += __shfl_xor(a1, off, 32);
  }
  if ((threadIdx.x & 31) == 0) {
    atomicAdd(&out[0], a0);
    if (edim > 1) atomicAdd(&out[1], a1);
  }
}

// k[h*edim+j] = sum_c w_edge[(h*128+c)*edim + j] * att_edge[h*128+c]   (kvec pre-zeroed)
__global__ void edge_k_k(const float* __restrict__ w_edge, const float* __restrict__ att_edge,
                         float* __restrict__ kvec, int edim) {
  int t = threadIdx.x;  // 0..255 = h*128+c
  int h = t >> 7;
  float av = att_edge[t];
  for (int j = 0; j < edim; ++j) atomicAdd(&kvec[h * edim + j], w_edge[(size_t)t * edim + j] * av);
}

// ---------------- WMMA GEMM: Y[M,N] = Xh[M,K] * Wh[N,K]^T ----------------
// One wave computes a 16x64 tile (4 N-tiles, A-fragment reused 4x);
// 4 waves/block cover 4 consecutive M tiles.
__global__ void wmma_gemm_nt(const _Float16* __restrict__ Xh, const _Float16* __restrict__ Wh,
                             float* __restrict__ Y, int M, int N, int K) {
  const int lane   = threadIdx.x & 31;
  const int wave   = threadIdx.x >> 5;
  const int nTile4 = blockIdx.x;            // 64-column group
  const int mTile  = blockIdx.y * 4 + wave; // 16-row tile
  const int hi     = lane >> 4;             // lane half
  const int l16    = lane & 15;
  const int kHalf  = hi * 8;  // 16-bit A/B fragment layout: K 0..7/16..23 vs 8..15/24..31
  const _Float16* pa  = Xh + (size_t)(mTile * 16 + l16) * K + kHalf;
  const _Float16* pb0 = Wh + (size_t)(nTile4 * 64 + l16) * K + kHalf;
  v8f acc[4] = {};
  for (int k0 = 0; k0 < K; k0 += 32) {
    v8h alo = *(const v8h*)(pa + k0);
    v8h ahi = *(const v8h*)(pa + k0 + 16);
    v16h a = __builtin_shufflevector(alo, ahi, 0,1,2,3,4,5,6,7,8,9,10,11,12,13,14,15);
#pragma unroll
    for (int t = 0; t < 4; ++t) {
      const _Float16* pb = pb0 + (size_t)(t * 16) * K + k0;
      v8h blo = *(const v8h*)(pb);
      v8h bhi = *(const v8h*)(pb + 16);
      v16h b = __builtin_shufflevector(blo, bhi, 0,1,2,3,4,5,6,7,8,9,10,11,12,13,14,15);
      acc[t] = __builtin_amdgcn_wmma_f32_16x16x32_f16(false, a, false, b, (short)0, acc[t],
                                                      false, false);
    }
  }
  // C/D layout: VGPR v, lanes 0-15 -> row v, lanes 16-31 -> row v+8; col = lane&15
#pragma unroll
  for (int t = 0; t < 4; ++t) {
    float* out = Y + (size_t)(mTile * 16 + hi * 8) * N + nTile4 * 64 + t * 16 + l16;
#pragma unroll
    for (int v = 0; v < 8; ++v) out[(size_t)v * N] = acc[t][v];
  }
}

// ---------------- per-node logits: out[n,h] = dot(x[n, h*128 : h*128+128], att[h]) ----------------
__global__ void node_logits_k(const float* __restrict__ x, const float* __restrict__ att,
                              float* __restrict__ out, int N) {
  int lane = threadIdx.x & 31;
  int wid  = ((int)blockIdx.x * blockDim.x + threadIdx.x) >> 5;
  if (wid >= N) return;
  const float* row = x + (size_t)wid * 256;
  float s0 = 0.f, s1 = 0.f;
#pragma unroll
  for (int i = 0; i < 8; ++i) {
    int idx = lane + 32 * i;
    float v = row[idx] * att[idx];
    if (i < 4) s0 += v; else s1 += v;
  }
  for (int off = 16; off > 0; off >>= 1) {
    s0 += __shfl_xor(s0, off, 32);
    s1 += __shfl_xor(s1, off, 32);
  }
  if (lane == 0) { out[wid * 2] = s0; out[wid * 2 + 1] = s1; }
}

// ---------------- edge attention logits + leaky(0.2) + segment max ----------------
__global__ void edge_attn_max_k(const int* __restrict__ src, const int* __restrict__ dst,
                                const float* __restrict__ as_, const float* __restrict__ ad_,
                                const float* __restrict__ ea, const float* __restrict__ kvec,
                                int edim, float* __restrict__ alpha, unsigned* __restrict__ m, int E) {
  int e = blockIdx.x * blockDim.x + threadIdx.x;
  if (e >= E) return;
  int s = src[e], d = dst[e];
  float et0 = 0.f, et1 = 0.f;
  for (int j = 0; j < edim; ++j) {
    float v = ea[(size_t)e * edim + j];
    et0 += v * kvec[j];
    et1 += v * kvec[edim + j];
  }
  float a0 = as_[s * 2]     + ad_[d * 2]     + et0;
  float a1 = as_[s * 2 + 1] + ad_[d * 2 + 1] + et1;
  a0 = a0 >= 0.f ? a0 : 0.2f * a0;
  a1 = a1 >= 0.f ? a1 : 0.2f * a1;
  alpha[(size_t)e * 2]     = a0;
  alpha[(size_t)e * 2 + 1] = a1;
  atomicMax(&m[d * 2],     encf(a0));
  atomicMax(&m[d * 2 + 1], encf(a1));
}

// self-loop edges (src=dst=n), edge attr = mean of original edge attrs
__global__ void selfloop_attn_max_k(const float* __restrict__ as_, const float* __restrict__ ad_,
                                    const float* __restrict__ meaAcc, float invE,
                                    const float* __restrict__ kvec, int edim,
                                    float* __restrict__ alphaLoop, unsigned* __restrict__ m, int N) {
  int n = blockIdx.x * blockDim.x + threadIdx.x;
  if (n >= N) return;
  float et0 = 0.f, et1 = 0.f;
  for (int j = 0; j < edim; ++j) {
    float me = meaAcc[j] * invE;
    et0 += me * kvec[j];
    et1 += me * kvec[edim + j];
  }
  float a0 = as_[n * 2]     + ad_[n * 2]     + et0;
  float a1 = as_[n * 2 + 1] + ad_[n * 2 + 1] + et1;
  a0 = a0 >= 0.f ? a0 : 0.2f * a0;
  a1 = a1 >= 0.f ? a1 : 0.2f * a1;
  alphaLoop[(size_t)n * 2]     = a0;
  alphaLoop[(size_t)n * 2 + 1] = a1;
  atomicMax(&m[n * 2],     encf(a0));
  atomicMax(&m[n * 2 + 1], encf(a1));
}

// ---------------- exp(a - m[dst]) and segment sum; dst==nullptr => self-loops ----------------
__global__ void edge_exp_sum_k(const int* __restrict__ dst, float* __restrict__ alpha,
                               const unsigned* __restrict__ m, float* __restrict__ ssum, int E) {
  int e = blockIdx.x * blockDim.x + threadIdx.x;
  if (e >= E) return;
  int d = dst ? dst[e] : e;
  float e0 = __expf(alpha[(size_t)e * 2]     - decf(m[d * 2]));
  float e1 = __expf(alpha[(size_t)e * 2 + 1] - decf(m[d * 2 + 1]));
  alpha[(size_t)e * 2]     = e0;
  alpha[(size_t)e * 2 + 1] = e1;
  atomicAdd(&ssum[d * 2],     e0);
  atomicAdd(&ssum[d * 2 + 1], e1);
}

// ---------------- alpha[e,h] /= (s[dst,h] + eps)  (1 thread/edge) ----------------
__global__ void normalize_k(const int* __restrict__ dst, float* __restrict__ alpha,
                            const float* __restrict__ ssum, int E) {
  int e = blockIdx.x * blockDim.x + threadIdx.x;
  if (e >= E) return;
  int d = dst ? dst[e] : e;
  alpha[(size_t)e * 2]     /= (ssum[d * 2]     + 1e-16f);
  alpha[(size_t)e * 2 + 1] /= (ssum[d * 2 + 1] + 1e-16f);
}

// ---------------- weighted scatter: acc[dst,h,c] += alpha * xs[src,h,c]; 64 threads/edge ----------------
__global__ void edge_scatter_k(const int* __restrict__ src, const int* __restrict__ dst,
                               const float* __restrict__ alpha,
                               const float* __restrict__ xs, float* __restrict__ acc, int E) {
  long gid = (long)blockIdx.x * blockDim.x + threadIdx.x;
  int e = (int)(gid >> 6);
  int t = (int)(gid & 63);
  if (e >= E) return;
  int s = src ? src[e] : e;
  int d = dst ? dst[e] : e;
  int base = t * 4;            // 0..252, within one head (128 % 4 == 0)
  int h = base >> 7;
  float w = alpha[(size_t)e * 2 + h];
  float4 xv = *(const float4*)(xs + (size_t)s * 256 + base);
  float* o = acc + (size_t)d * 256 + base;
  atomicAdd(o + 0, w * xv.x);
  atomicAdd(o + 1, w * xv.y);
  atomicAdd(o + 2, w * xv.z);
  atomicAdd(o + 3, w * xv.w);
}

// ---------------- hetero combine ----------------
__global__ void combine_patient_k(const float* __restrict__ app, const float* __restrict__ bpp,
                                  const float* __restrict__ afp, const float* __restrict__ bfp,
                                  float* __restrict__ xp, int N) {
  int i = blockIdx.x * blockDim.x + threadIdx.x;
  if (i >= N * 128) return;
  int n = i >> 7, c = i & 127;
  float vpp = 0.5f * (app[(size_t)n * 256 + c] + app[(size_t)n * 256 + 128 + c]) + bpp[c];
  float vfp = 0.5f * (afp[(size_t)n * 256 + c] + afp[(size_t)n * 256 + 128 + c]) + bfp[c];
  float v = 0.5f * (vpp + vfp);
  xp[i] = v >= 0.f ? v : 0.01f * v;
}

__global__ void combine_feature_k(const float* __restrict__ aff, const float* __restrict__ bff,
                                  float* __restrict__ xf, int N) {
  int i = blockIdx.x * blockDim.x + threadIdx.x;
  if (i >= N * 128) return;
  int n = i >> 7, c = i & 127;
  float v = 0.5f * (aff[(size_t)n * 256 + c] + aff[(size_t)n * 256 + 128 + c]) + bff[c];
  xf[i] = v >= 0.f ? v : 0.01f * v;
}

// ---------------- final linear: out[n,o] = dot(xp[n,:128], lw[o,:]) + lb[o] ----------------
__global__ void final_linear_k(const float* __restrict__ xp, const float* __restrict__ lw,
                               const float* __restrict__ lb, float* __restrict__ out, int N) {
  int lane = threadIdx.x & 31;
  int wid  = ((int)blockIdx.x * blockDim.x + threadIdx.x) >> 5;
  if (wid >= N) return;
  const float* row = xp + (size_t)wid * 128;
  float s0 = 0.f, s1 = 0.f;
#pragma unroll
  for (int i = 0; i < 4; ++i) {
    int c = lane + 32 * i;
    float v = row[c];
    s0 += v * lw[c];
    s1 += v * lw[128 + c];
  }
  for (int off = 16; off > 0; off >>= 1) {
    s0 += __shfl_xor(s0, off, 32);
    s1 += __shfl_xor(s1, off, 32);
  }
  if (lane == 0) { out[wid * 2] = s0 + lb[0]; out[wid * 2 + 1] = s1 + lb[1]; }
}

// ==================== host side ====================
namespace {

struct WS {
  float *xs_pp, *xs_ff, *xs_fp, *xd_fp;
  float *acc_pp, *acc_ff, *acc_fp;
  float *xp, *xf;
  float *as_pp, *ad_pp, *as_ff, *ad_ff, *as_fp, *ad_fp;
  float *s_pp, *s_ff, *s_fp;
  float *alpha_pp, *alpha_ff, *alpha_fp;
  float *meanea_pp, *meanea_ff, *kvec;
  unsigned *m_pp, *m_ff, *m_fp;
  _Float16 *xph, *xfh, *wh;
};

inline void fillu(hipStream_t st, void* p, unsigned v, long n) {
  fill_u32_k<<<(unsigned)((n + 255) / 256), 256, 0, st>>>((unsigned*)p, v, n);
}
inline void cvt(hipStream_t st, const float* s, _Float16* d, long n) {
  f32_to_f16_k<<<(unsigned)((n + 255) / 256), 256, 0, st>>>(s, d, n);
}
// Y[M,256] = Ah[M,K] * W[256,K]^T  (W converted to f16 into ws.wh first)
inline void gemm(hipStream_t st, WS& w, const _Float16* Ah, int M, int K, const float* W, float* Y) {
  cvt(st, W, w.wh, (long)256 * K);
  dim3 g(256 / 64, M / 64);
  wmma_gemm_nt<<<g, 128, 0, st>>>(Ah, w.wh, Y, M, 256, K);
}

struct RelSelfP { const float *att_dst, *att_edge, *att_src, *bias, *w_edge, *w_src; };
struct RelFpP   { const float *att_dst, *att_src, *bias, *w_dst, *w_src; };

void relation_self(hipStream_t st, WS& w, const _Float16* Ah, int N, int K, const RelSelfP& P,
                   int edim, const int* src, const int* dst, const float* ea, int E,
                   const float* meaAcc, float invE,
                   float* xs, float* as_, float* ad_, unsigned* m, float* s, float* alpha, float* acc) {
  gemm(st, w, Ah, N, K, P.w_src, xs);
  node_logits_k<<<(N * 32 + 255) / 256, 256, 0, st>>>(xs, P.att_src, as_, N);
  node_logits_k<<<(N * 32 + 255) / 256, 256, 0, st>>>(xs, P.att_dst, ad_, N);
  fillu(st, w.kvec, 0u, 4);
  edge_k_k<<<1, 256, 0, st>>>(P.w_edge, P.att_edge, w.kvec, edim);
  fillu(st, m, ENC_NEG_INF, (long)N * 2);
  fillu(st, s, 0u, (long)N * 2);
  fillu(st, acc, 0u, (long)N * 256);
  edge_attn_max_k<<<(E + 255) / 256, 256, 0, st>>>(src, dst, as_, ad_, ea, w.kvec, edim, alpha, m, E);
  selfloop_attn_max_k<<<(N + 255) / 256, 256, 0, st>>>(as_, ad_, meaAcc, invE, w.kvec, edim,
                                                       alpha + (size_t)E * 2, m, N);
  edge_exp_sum_k<<<(E + 255) / 256, 256, 0, st>>>(dst, alpha, m, s, E);
  edge_exp_sum_k<<<(N + 255) / 256, 256, 0, st>>>(nullptr, alpha + (size_t)E * 2, m, s, N);
  normalize_k<<<(E + 255) / 256, 256, 0, st>>>(dst, alpha, s, E);
  normalize_k<<<(N + 255) / 256, 256, 0, st>>>(nullptr, alpha + (size_t)E * 2, s, N);
  long te = (long)E * 64, tn = (long)N * 64;
  edge_scatter_k<<<(unsigned)((te + 255) / 256), 256, 0, st>>>(src, dst, alpha, xs, acc, E);
  edge_scatter_k<<<(unsigned)((tn + 255) / 256), 256, 0, st>>>(nullptr, nullptr,
                                                               alpha + (size_t)E * 2, xs, acc, N);
}

void relation_fp(hipStream_t st, WS& w, const _Float16* Afh, int Kf, const _Float16* Aph, int Kp,
                 const RelFpP& P, const int* src, const int* dst, int E) {
  gemm(st, w, Afh, NF, Kf, P.w_src, w.xs_fp);
  gemm(st, w, Aph, NP, Kp, P.w_dst, w.xd_fp);
  node_logits_k<<<(NF * 32 + 255) / 256, 256, 0, st>>>(w.xs_fp, P.att_src, w.as_fp, NF);
  node_logits_k<<<(NP * 32 + 255) / 256, 256, 0, st>>>(w.xd_fp, P.att_dst, w.ad_fp, NP);
  fillu(st, w.m_fp, ENC_NEG_INF, (long)NP * 2);
  fillu(st, w.s_fp, 0u, (long)NP * 2);
  fillu(st, w.acc_fp, 0u, (long)NP * 256);
  edge_attn_max_k<<<(E + 255) / 256, 256, 0, st>>>(src, dst, w.as_fp, w.ad_fp, nullptr, nullptr, 0,
                                                   w.alpha_fp, w.m_fp, E);
  edge_exp_sum_k<<<(E + 255) / 256, 256, 0, st>>>(dst, w.alpha_fp, w.m_fp, w.s_fp, E);
  normalize_k<<<(E + 255) / 256, 256, 0, st>>>(dst, w.alpha_fp, w.s_fp, E);
  long te = (long)E * 64;
  edge_scatter_k<<<(unsigned)((te + 255) / 256), 256, 0, st>>>(src, dst, w.alpha_fp,
                                                               w.xs_fp, w.acc_fp, E);
}

}  // namespace

extern "C" void kernel_launch(void* const* d_in, const int* in_sizes, int n_in,
                              void* d_out, int out_size, void* d_ws, size_t ws_size,
                              hipStream_t stream) {
  (void)in_sizes; (void)n_in; (void)out_size; (void)ws_size;
  const float* x_patient = (const float*)d_in[0];
  const float* x_feature = (const float*)d_in[1];
  const int*   ei_pp     = (const int*)d_in[2];
  const int*   ei_ff     = (const int*)d_in[3];
  const int*   ei_fp     = (const int*)d_in[4];
  const float* ea_pp     = (const float*)d_in[5];
  const float* ea_ff     = (const float*)d_in[6];
  auto F = [&](int i) { return (const float*)d_in[i]; };
  // params pytree flatten (sorted dict keys): layers[0]{ff,fp,pp}, layers[1]{ff,fp,pp}, lin_b, lin_w
  RelSelfP ff[2] = { {F(7),F(8),F(9),F(10),F(11),F(12)},   {F(24),F(25),F(26),F(27),F(28),F(29)} };
  RelFpP   fp[2] = { {F(13),F(14),F(15),F(16),F(17)},      {F(30),F(31),F(32),F(33),F(34)} };
  RelSelfP pp[2] = { {F(18),F(19),F(20),F(21),F(22),F(23)},{F(35),F(36),F(37),F(38),F(39),F(40)} };
  const float* lin_b = F(41);
  const float* lin_w = F(42);

  // workspace bump allocator (256B aligned)
  char* base = (char*)d_ws;
  size_t off = 0;
  auto alloc = [&](size_t bytes) -> void* {
    void* p = base + off;
    off = (off + bytes + 255) & ~(size_t)255;
    return p;
  };
  WS w;
  w.xs_pp   = (float*)alloc((size_t)NP * 256 * 4);
  w.xs_ff   = (float*)alloc((size_t)NF * 256 * 4);
  w.xs_fp   = (float*)alloc((size_t)NF * 256 * 4);
  w.xd_fp   = (float*)alloc((size_t)NP * 256 * 4);
  w.acc_pp  = (float*)alloc((size_t)NP * 256 * 4);
  w.acc_ff  = (float*)alloc((size_t)NF * 256 * 4);
  w.acc_fp  = (float*)alloc((size_t)NP * 256 * 4);
  w.xp      = (float*)alloc((size_t)NP * 128 * 4);
  w.xf      = (float*)alloc((size_t)NF * 128 * 4);
  w.as_pp   = (float*)alloc((size_t)NP * 2 * 4);
  w.ad_pp   = (float*)alloc((size_t)NP * 2 * 4);
  w.as_ff   = (float*)alloc((size_t)NF * 2 * 4);
  w.ad_ff   = (float*)alloc((size_t)NF * 2 * 4);
  w.as_fp   = (float*)alloc((size_t)NF * 2 * 4);
  w.ad_fp   = (float*)alloc((size_t)NP * 2 * 4);
  w.s_pp    = (float*)alloc((size_t)NP * 2 * 4);
  w.s_ff    = (float*)alloc((size_t)NF * 2 * 4);
  w.s_fp    = (float*)alloc((size_t)NP * 2 * 4);
  w.alpha_pp = (float*)alloc((size_t)(EPP + NP) * 2 * 4);
  w.alpha_ff = (float*)alloc((size_t)(EFF + NF) * 2 * 4);
  w.alpha_fp = (float*)alloc((size_t)EFP * 2 * 4);
  w.meanea_pp = (float*)alloc(2 * 4);
  w.meanea_ff = (float*)alloc(2 * 4);
  w.kvec      = (float*)alloc(4 * 4);
  w.m_pp    = (unsigned*)alloc((size_t)NP * 2 * 4);
  w.m_ff    = (unsigned*)alloc((size_t)NF * 2 * 4);
  w.m_fp    = (unsigned*)alloc((size_t)NP * 2 * 4);
  w.xph     = (_Float16*)alloc((size_t)NP * 256 * 2);
  w.xfh     = (_Float16*)alloc((size_t)NF * 128 * 2);
  w.wh      = (_Float16*)alloc((size_t)256 * 256 * 2);

  // mean edge attrs (sums; scaled by 1/E when used)
  fillu(stream, w.meanea_pp, 0u, 2);
  fillu(stream, w.meanea_ff, 0u, 2);
  mean_accum_k<<<256, 256, 0, stream>>>(ea_pp, w.meanea_pp, EPP, 1);
  mean_accum_k<<<256, 256, 0, stream>>>(ea_ff, w.meanea_ff, EFF, 2);

  const float* xpcur = x_patient; int dp = 256;
  const float* xfcur = x_feature; int df = 64;
  const int* spp = ei_pp, *dpp = ei_pp + EPP;
  const int* sff = ei_ff, *dff = ei_ff + EFF;
  const int* sfp = ei_fp, *dfp = ei_fp + EFP;

  for (int l = 0; l < 2; ++l) {
    cvt(stream, xpcur, w.xph, (long)NP * dp);
    cvt(stream, xfcur, w.xfh, (long)NF * df);

    relation_self(stream, w, w.xph, NP, dp, pp[l], 1, spp, dpp, ea_pp, EPP,
                  w.meanea_pp, 1.f / EPP,
                  w.xs_pp, w.as_pp, w.ad_pp, w.m_pp, w.s_pp, w.alpha_pp, w.acc_pp);
    relation_self(stream, w, w.xfh, NF, df, ff[l], 2, sff, dff, ea_ff, EFF,
                  w.meanea_ff, 1.f / EFF,
                  w.xs_ff, w.as_ff, w.ad_ff, w.m_ff, w.s_ff, w.alpha_ff, w.acc_ff);
    relation_fp(stream, w, w.xfh, df, w.xph, dp, fp[l], sfp, dfp, EFP);

    combine_patient_k<<<(NP * 128 + 255) / 256, 256, 0, stream>>>(
        w.acc_pp, pp[l].bias, w.acc_fp, fp[l].bias, w.xp, NP);
    combine_feature_k<<<(NF * 128 + 255) / 256, 256, 0, stream>>>(
        w.acc_ff, ff[l].bias, w.xf, NF);

    xpcur = w.xp; dp = 128;
    xfcur = w.xf; df = 128;
  }

  final_linear_k<<<(NP * 32 + 255) / 256, 256, 0, stream>>>(w.xp, lin_w, lin_b, (float*)d_out, NP);
}
